// BasedXLLowPLinear_52939766890666
// MI455X (gfx1250) — matrix operand order repaired and checked by reading, not compiled, not run on hardware
//
#include <hip/hip_runtime.h>
#include <hip/hip_bf16.h>

typedef __attribute__((ext_vector_type(16))) __bf16 bf16x16;
typedef __attribute__((ext_vector_type(8)))  __bf16 bf16x8;
typedef __attribute__((ext_vector_type(8)))  float  f32x8;

union FragBF16 { bf16x16 v; bf16x8 h[2]; };

#define TILE_M 128
#define TILE_N 128
#define TILE_K 32
#define LDS_STRIDE 40   // halves per row: 32 data + 8 pad, keeps 16B alignment

__device__ __forceinline__ unsigned rne_hi(unsigned u) {
    // round-to-nearest-even fp32 -> bf16 (bit trick), returns low 16 bits
    return (u + 0x7FFFu + ((u >> 16) & 1u)) >> 16;
}

// ---------------------------------------------------------------------------
// Pass 1a: elementwise split X (fp32, row-major [M][K]) -> Xh, Xl (bf16, same)
// Bandwidth-bound; exact RNE like jnp astype(bfloat16).
// ---------------------------------------------------------------------------
__global__ __launch_bounds__(256)
void split_rows(const float* __restrict__ src,
                __bf16* __restrict__ hi, __bf16* __restrict__ lo, size_t n)
{
    size_t i = ((size_t)blockIdx.x * blockDim.x + threadIdx.x) * 8;
    if (i >= n) return;
    const float4* p = (const float4*)(src + i);
    float4 r0 = p[0], r1 = p[1];
    float f[8] = {r0.x, r0.y, r0.z, r0.w, r1.x, r1.y, r1.z, r1.w};
    unsigned hp[4], lp[4];
    #pragma unroll
    for (int t = 0; t < 4; t++) {
        unsigned u0 = __float_as_uint(f[2*t]);
        unsigned u1 = __float_as_uint(f[2*t+1]);
        unsigned h0 = rne_hi(u0), h1 = rne_hi(u1);
        hp[t] = h0 | (h1 << 16);
        float l0 = f[2*t]   - __uint_as_float(h0 << 16);
        float l1 = f[2*t+1] - __uint_as_float(h1 << 16);
        lp[t] = rne_hi(__float_as_uint(l0)) | (rne_hi(__float_as_uint(l1)) << 16);
    }
    *(uint4*)(hi + i) = make_uint4(hp[0], hp[1], hp[2], hp[3]);
    *(uint4*)(lo + i) = make_uint4(lp[0], lp[1], lp[2], lp[3]);
}

// ---------------------------------------------------------------------------
// Pass 1b: split + transpose W (fp32, [K][N]) -> WhT, WlT (bf16, [N][K]).
// Output is K-contiguous so the GEMM's B fragments are straight b128 copies.
// ---------------------------------------------------------------------------
__global__ __launch_bounds__(256)
void split_transpose(const float* __restrict__ W,
                     __bf16* __restrict__ hiT, __bf16* __restrict__ loT,
                     int K, int N)
{
    __shared__ __bf16 sh[64][72];
    __shared__ __bf16 sl[64][72];
    int n0 = blockIdx.x * 64;
    int k0 = blockIdx.y * 64;
    int tid = threadIdx.x;

    int krow = tid >> 2;
    int nc   = (tid & 3) << 4;
    const float4* p = (const float4*)(W + (size_t)(k0 + krow) * N + n0 + nc);
    float4 r[4] = {p[0], p[1], p[2], p[3]};
    const float* f = (const float*)r;
    #pragma unroll
    for (int t = 0; t < 16; t++) {
        unsigned u = __float_as_uint(f[t]);
        unsigned h = rne_hi(u);
        unsigned l = rne_hi(__float_as_uint(f[t] - __uint_as_float(h << 16)));
        sh[krow][nc + t] = __builtin_bit_cast(__bf16, (unsigned short)h);
        sl[krow][nc + t] = __builtin_bit_cast(__bf16, (unsigned short)l);
    }
    __syncthreads();

    int nrow = tid >> 2;
    int kc   = (tid & 3) << 4;
    bf16x8 vh0, vh1, vl0, vl1;
    #pragma unroll
    for (int t = 0; t < 8; t++) {
        vh0[t] = sh[kc + t][nrow];
        vh1[t] = sh[kc + 8 + t][nrow];
        vl0[t] = sl[kc + t][nrow];
        vl1[t] = sl[kc + 8 + t][nrow];
    }
    __bf16* oh = hiT + (size_t)(n0 + nrow) * K + k0 + kc;
    __bf16* ol = loT + (size_t)(n0 + nrow) * K + k0 + kc;
    *(bf16x8*)oh = vh0; *(bf16x8*)(oh + 8) = vh1;
    *(bf16x8*)ol = vl0; *(bf16x8*)(ol + 8) = vl1;
}

// ---------------------------------------------------------------------------
// Pass 2: the WMMA GEMM.  PRESPLIT=true reads pre-split bf16 (zero conversion
// VALU in the main loop); PRESPLIT=false is the fused fallback when ws is
// too small, using cheap truncation splits (v_perm packing).
// ---------------------------------------------------------------------------
template <bool PRESPLIT>
__global__ __launch_bounds__(256)
void cascaded_gemm(const float* __restrict__ X, const float* __restrict__ W,
                   const __bf16* __restrict__ Xh, const __bf16* __restrict__ Xl,
                   const __bf16* __restrict__ WhT, const __bf16* __restrict__ WlT,
                   const float* __restrict__ Bias, float* __restrict__ Out,
                   int M, int N, int K, int grid_n)
{
    // [buffer][plane: 0=A_hi 1=A_lo 2=B_hi(T) 3=B_lo(T)][128 rows * 40 halves]
    __shared__ __align__(16) __bf16 lds[2][4][TILE_M * LDS_STRIDE];

    // 8x8 supertile swizzle: 16.8MB A-panel + 16.8MB B-panel stay hot in L2
    int bid        = blockIdx.x;
    int super_cols = grid_n >> 3;
    int super      = bid >> 6;
    int inner      = bid & 63;
    int bm = ((super / super_cols) << 3) + (inner >> 3);
    int bn = ((super % super_cols) << 3) + (inner & 7);
    int m0 = bm * TILE_M;
    int n0 = bn * TILE_N;

    int tid  = threadIdx.x;
    int lane = tid & 31;
    int wave = tid >> 5;
    int wm   = (wave >> 2) << 6;   // 0 or 64
    int wn   = (wave & 3)  << 5;   // 0..96

    // presplit staging coords: 16 halves per thread per plane
    int crow  = tid >> 1;          // 0..127 (A: M-row, B: N-row)
    int ckoff = (tid & 1) << 4;    // 0 or 16
    // fused staging coords
    int arow = tid >> 1, acol = (tid & 1) << 4;
    int brow = tid >> 3, bcol = (tid & 7) << 4;

    f32x8 acc[4][2];
    #pragma unroll
    for (int i = 0; i < 4; i++)
        #pragma unroll
        for (int j = 0; j < 2; j++)
            acc[i][j] = {};

    bf16x8 pr[8];          // presplit staged regs (32 VGPRs)
    float4 ar[4], br[4];   // fused staged regs

    auto loadTiles = [&](int k0) {
        if (PRESPLIT) {
            const __bf16* pAh = Xh  + (size_t)(m0 + crow) * K + k0 + ckoff;
            const __bf16* pAl = Xl  + (size_t)(m0 + crow) * K + k0 + ckoff;
            const __bf16* pBh = WhT + (size_t)(n0 + crow) * K + k0 + ckoff;
            const __bf16* pBl = WlT + (size_t)(n0 + crow) * K + k0 + ckoff;
            pr[0] = *(const bf16x8*)(pAh); pr[1] = *(const bf16x8*)(pAh + 8);
            pr[2] = *(const bf16x8*)(pAl); pr[3] = *(const bf16x8*)(pAl + 8);
            pr[4] = *(const bf16x8*)(pBh); pr[5] = *(const bf16x8*)(pBh + 8);
            pr[6] = *(const bf16x8*)(pBl); pr[7] = *(const bf16x8*)(pBl + 8);
        } else {
            const float4* pa = (const float4*)(X + (size_t)(m0 + arow) * K + k0 + acol);
            ar[0] = pa[0]; ar[1] = pa[1]; ar[2] = pa[2]; ar[3] = pa[3];
            const float4* pb = (const float4*)(W + (size_t)(k0 + brow) * N + n0 + bcol);
            br[0] = pb[0]; br[1] = pb[1]; br[2] = pb[2]; br[3] = pb[3];
        }
    };

    auto storeTiles = [&](int buf) {
        if (PRESPLIT) {
            int off = crow * LDS_STRIDE + ckoff;
            *(bf16x8*)&lds[buf][0][off]     = pr[0];
            *(bf16x8*)&lds[buf][0][off + 8] = pr[1];
            *(bf16x8*)&lds[buf][1][off]     = pr[2];
            *(bf16x8*)&lds[buf][1][off + 8] = pr[3];
            *(bf16x8*)&lds[buf][2][off]     = pr[4];
            *(bf16x8*)&lds[buf][2][off + 8] = pr[5];
            *(bf16x8*)&lds[buf][3][off]     = pr[6];
            *(bf16x8*)&lds[buf][3][off + 8] = pr[7];
        } else {
            // A: truncation split, v_perm-packed pairs, vector LDS stores
            const float* f = (const float*)ar;
            unsigned hp[8], lp[8];
            #pragma unroll
            for (int t = 0; t < 8; t++) {
                unsigned u0 = __float_as_uint(f[2*t]);
                unsigned u1 = __float_as_uint(f[2*t+1]);
                hp[t] = __builtin_amdgcn_perm(u1, u0, 0x07060302u); // {hi16(u1),hi16(u0)}
                float l0 = f[2*t]   - __uint_as_float(u0 & 0xFFFF0000u);
                float l1 = f[2*t+1] - __uint_as_float(u1 & 0xFFFF0000u);
                lp[t] = __builtin_amdgcn_perm(__float_as_uint(l1),
                                              __float_as_uint(l0), 0x07060302u);
            }
            unsigned* Ah = (unsigned*)&lds[buf][0][arow * LDS_STRIDE + acol];
            unsigned* Al = (unsigned*)&lds[buf][1][arow * LDS_STRIDE + acol];
            *(uint4*)(Ah)     = make_uint4(hp[0], hp[1], hp[2], hp[3]);
            *(uint4*)(Ah + 4) = make_uint4(hp[4], hp[5], hp[6], hp[7]);
            *(uint4*)(Al)     = make_uint4(lp[0], lp[1], lp[2], lp[3]);
            *(uint4*)(Al + 4) = make_uint4(lp[4], lp[5], lp[6], lp[7]);
            // B: truncation split with transposed scatter stores
            const float* g = (const float*)br;
            __bf16* Bh = lds[buf][2];
            __bf16* Bl = lds[buf][3];
            #pragma unroll
            for (int t = 0; t < 16; t++) {
                unsigned u = __float_as_uint(g[t]);
                float lf = g[t] - __uint_as_float(u & 0xFFFF0000u);
                unsigned lu = __float_as_uint(lf);
                Bh[(bcol + t) * LDS_STRIDE + brow] =
                    __builtin_bit_cast(__bf16, (unsigned short)(u >> 16));
                Bl[(bcol + t) * LDS_STRIDE + brow] =
                    __builtin_bit_cast(__bf16, (unsigned short)(lu >> 16));
            }
        }
    };

    auto compute = [&](int buf) {
        const __bf16* Ah = lds[buf][0];
        const __bf16* Al = lds[buf][1];
        const __bf16* Bh = lds[buf][2];
        const __bf16* Bl = lds[buf][3];
        int row = lane & 15;
        int akf = (lane >> 4) << 3;   // A: lane half -> K 0-7/16-23 vs 8-15/24-31
        int bkf = (lane >> 4) << 3;   // B: lane half -> K 0-7/16-23 vs 8-15/24-31

        FragBF16 ah[4], al[4];
        #pragma unroll
        for (int i = 0; i < 4; i++) {
            const __bf16* p = &Ah[(wm + i * 16 + row) * LDS_STRIDE + akf];
            ah[i].h[0] = *(const bf16x8*)(p);
            ah[i].h[1] = *(const bf16x8*)(p + 16);
            const __bf16* q = &Al[(wm + i * 16 + row) * LDS_STRIDE + akf];
            al[i].h[0] = *(const bf16x8*)(q);
            al[i].h[1] = *(const bf16x8*)(q + 16);
        }
        #pragma unroll
        for (int j = 0; j < 2; j++) {
            FragBF16 bh, bl;
            const __bf16* p = &Bh[(wn + j * 16 + row) * LDS_STRIDE + bkf];
            bh.h[0] = *(const bf16x8*)(p);
            bh.h[1] = *(const bf16x8*)(p + 16);
            const __bf16* q = &Bl[(wn + j * 16 + row) * LDS_STRIDE + bkf];
            bl.h[0] = *(const bf16x8*)(q);
            bl.h[1] = *(const bf16x8*)(q + 16);
            #pragma unroll
            for (int i = 0; i < 4; i++) {
                acc[i][j] = __builtin_amdgcn_wmma_f32_16x16x32_bf16(
                    false, ah[i].v, false, bh.v, (short)0, acc[i][j], false, false);
                acc[i][j] = __builtin_amdgcn_wmma_f32_16x16x32_bf16(
                    false, ah[i].v, false, bl.v, (short)0, acc[i][j], false, false);
                acc[i][j] = __builtin_amdgcn_wmma_f32_16x16x32_bf16(
                    false, al[i].v, false, bh.v, (short)0, acc[i][j], false, false);
            }
        }
    };

    // prologue
    loadTiles(0);
    storeTiles(0);
    __syncthreads();

    int nk = K / TILE_K;
    for (int kt = 0; kt < nk; kt++) {
        if (PRESPLIT) {
            if (kt + 2 < nk) {
                __builtin_prefetch(Xh  + (size_t)(m0 + crow) * K + (kt + 2) * TILE_K, 0, 1);
                __builtin_prefetch(WhT + (size_t)(n0 + crow) * K + (kt + 2) * TILE_K, 0, 1);
            }
        }
        if (kt + 1 < nk) loadTiles((kt + 1) * TILE_K);
        compute(kt & 1);
        if (kt + 1 < nk) storeTiles((kt + 1) & 1);
        __syncthreads();
    }

    // epilogue: C layout => lane half selects rows v / v+8
    #pragma unroll
    for (int i = 0; i < 4; i++) {
        int mglob = m0 + wm + i * 16 + ((lane >> 4) << 3);
        #pragma unroll
        for (int j = 0; j < 2; j++) {
            int nglob = n0 + wn + j * 16 + (lane & 15);
            float bv = Bias[nglob];
            float* o = Out + (size_t)mglob * N + nglob;
            #pragma unroll
            for (int v = 0; v < 8; v++)
                o[(size_t)v * N] = acc[i][j][v] + bv;
        }
    }
}

extern "C" void kernel_launch(void* const* d_in, const int* in_sizes, int n_in,
                              void* d_out, int out_size, void* d_ws, size_t ws_size,
                              hipStream_t stream) {
    const float* X    = (const float*)d_in[0];   // [M, K] fp32
    const float* W    = (const float*)d_in[1];   // [K, N] fp32
    const float* Bias = (const float*)d_in[2];   // [N] fp32
    float* Out        = (float*)d_out;           // [M, N] fp32

    long N = in_sizes[2];               // 16384
    long K = in_sizes[1] / N;           // 4096
    long M = (long)in_sizes[0] / K;     // 8192

    int grid_m = (int)(M / TILE_M);     // 64
    int grid_n = (int)(N / TILE_N);     // 128
    dim3 grid((unsigned)(grid_m * grid_n)), block(256);

    size_t szX  = (size_t)M * (size_t)K;
    size_t szW  = (size_t)K * (size_t)N;
    size_t need = (szX + szW) * 2 * sizeof(unsigned short);

    if (d_ws && ws_size >= need) {
        __bf16* Xh  = (__bf16*)d_ws;
        __bf16* Xl  = Xh + szX;
        __bf16* WhT = Xl + szX;
        __bf16* WlT = WhT + szW;
        unsigned nbx = (unsigned)((szX / 8 + 255) / 256);
        split_rows<<<dim3(nbx), dim3(256), 0, stream>>>(X, Xh, Xl, szX);
        dim3 tg((unsigned)(N / 64), (unsigned)(K / 64));
        split_transpose<<<tg, dim3(256), 0, stream>>>(W, WhT, WlT, (int)K, (int)N);
        cascaded_gemm<true><<<grid, block, 0, stream>>>(
            X, W, Xh, Xl, WhT, WlT, Bias, Out, (int)M, (int)N, (int)K, grid_n);
    } else {
        cascaded_gemm<false><<<grid, block, 0, stream>>>(
            X, W, nullptr, nullptr, nullptr, nullptr, Bias, Out,
            (int)M, (int)N, (int)K, grid_n);
    }
}
